// MinGRU_7541962572539
// MI455X (gfx1250) — matrix operand order repaired and checked by reading, not compiled, not run on hardware
//
#include <hip/hip_runtime.h>
#include <math.h>

// Problem shape (fixed by reference setup_inputs)
#define B_DIM 8
#define S_DIM 4096
#define D_DIM 1024
#define H_DIM 1024
#define M_TOTAL (B_DIM * S_DIM)   // 32768

// GEMM tiling: block = 256 threads = 8 waves (wave32).
// Block tile: 128(M) x 64(N), K-step 32 (bf16 WMMA K=32), double-buffered LDS.
// Waves arranged 4x2; each wave: 32x32 = 2x2 WMMA 16x16 tiles, for BOTH GEMMs.
#define BLK_M 128
#define BLK_N 64
#define BLK_K 32
#define NK    (D_DIM / BLK_K)     // 32 k-steps
#define LDA_U32 20                // A row stride in u32 (16 data + pad, 80B: 16B-aligned)
#define LDB_U32 20                // B row stride in u32 (16 kp + pad)

typedef __attribute__((ext_vector_type(16))) __bf16 v16bf;
typedef __attribute__((ext_vector_type(8)))  float  v8f;

// TDM descriptor vector types (probe-confirmed for ROCm 7.2 / clang-22)
typedef __attribute__((ext_vector_type(4))) unsigned tdm_v4u;
typedef __attribute__((ext_vector_type(8))) int      tdm_v8i;
typedef __attribute__((ext_vector_type(4))) int      tdm_v4i;

// pack two f32 into one u32 of bf16 pair {lo = even K, hi = odd K} (truncation)
static __device__ __forceinline__ unsigned pack_bf16(float lo, float hi_) {
  const unsigned ul = __builtin_bit_cast(unsigned, lo);
  const unsigned uh = __builtin_bit_cast(unsigned, hi_);
  return (ul >> 16) | (uh & 0xFFFF0000u);
}

// A fragment: u32s [0..3] = K kb..kb+7, [4..7] = K kb+8..kb+15 (contiguous, matches B)
static __device__ __forceinline__ v16bf load_frag_a(const unsigned* p) {
  union { int4 q[2]; v16bf v; } u;
  u.q[0] = *(const int4*)(p);
  u.q[1] = *(const int4*)(p + 4);
  return u.v;
}
// B fragment: 8 contiguous u32 (kp = hi*8 .. hi*8+7) at row n of the paired-transposed tile
static __device__ __forceinline__ v16bf load_frag_b(const unsigned* p) {
  union { int4 q[2]; v16bf v; } u;
  u.q[0] = *(const int4*)(p);
  u.q[1] = *(const int4*)(p + 4);
  return u.v;
}

static __device__ __forceinline__ float softplusf(float x) {
  return fmaxf(x, 0.0f) + log1pf(__expf(-fabsf(x)));   // stable softplus
}

// raw 32-bit LDS byte offset of a __shared__ object
static __device__ __forceinline__ unsigned lds_offset(const void* p) {
  return (unsigned)(unsigned long long)
      (__attribute__((address_space(3))) const char*)p;
}

// Issue a TDM 2-D tile load: tile_x contiguous f32 elements, tile_y rows,
// row stride = stride_elems f32, into LDS at lds_off (rows packed contiguously).
// D# bitfields per CDNA5 ISA (count=1, data_size=4B, type=2).
static __device__ __forceinline__ void tdm_load_2d(unsigned lds_off, const float* gptr,
                                                   int tile_x, int tile_y,
                                                   int stride_elems) {
  const unsigned long long ga = (unsigned long long)(size_t)gptr;
  tdm_v4u g0;
  g0[0] = 1u;                                            // count=1, no gather
  g0[1] = lds_off;                                       // lds_addr
  g0[2] = (unsigned)(ga & 0xFFFFFFFFull);                // global_addr[31:0]
  g0[3] = (unsigned)((ga >> 32) & 0x01FFFFFFull)         // global_addr[56:32]
          | (2u << 30);                                  // type=2 ("image")
  tdm_v8i g1;
  g1[0] = (int)(2u << 16);                               // data_size=2 -> 4 bytes
  g1[1] = (int)(((unsigned)tile_x & 0xFFFFu) << 16);     // tensor_dim0[15:0]
  g1[2] = (int)((((unsigned)tile_x >> 16) & 0xFFFFu)     // tensor_dim0[31:16]
          | (((unsigned)tile_y & 0xFFFFu) << 16));       // tensor_dim1[15:0]
  g1[3] = (int)((((unsigned)tile_y >> 16) & 0xFFFFu)     // tensor_dim1[31:16]
          | (((unsigned)tile_x & 0xFFFFu) << 16));       // tile_dim0
  g1[4] = (int)((unsigned)tile_y & 0xFFFFu);             // tile_dim1 (tile_dim2=0)
  g1[5] = stride_elems;                                  // tensor_dim0_stride[31:0]
  g1[6] = 0;                                             // stride hi / dim1_stride lo
  g1[7] = 0;
  const tdm_v4i gz = (tdm_v4i)0;
#if __clang_major__ >= 23
  const tdm_v8i gz8 = (tdm_v8i)0;
  __builtin_amdgcn_tensor_load_to_lds(g0, g1, gz, gz, gz8, 0);
#else
  __builtin_amdgcn_tensor_load_to_lds(g0, g1, gz, gz, 0);
#endif
}

// ---------------------------------------------------------------------------
// Kernel 1: fused dual GEMM (k = x@Wz+bz, t = x@Wh+bh) + log-space epilogue.
//   la (ws)    <- log_coeffs = -softplus(k)
//   lb (d_out) <- log_z + log_tilde_h = -softplus(-k) + log_g(t)
// LDS holds packed bf16; fragments are pure ds_load_b128, WMMA = bf16 16x16x32.
// ---------------------------------------------------------------------------
__global__ __launch_bounds__(256)
void mingru_gemm_kernel(const float* __restrict__ x,
                        const float* __restrict__ Wz,
                        const float* __restrict__ Wh,
                        const float* __restrict__ bz,
                        const float* __restrict__ bh,
                        float* __restrict__ la,
                        float* __restrict__ lb) {
  // A: [row][k/2] packed pairs, row-major.  B: [n][kp] paired-transposed.
  __shared__ unsigned sAp [2][BLK_M][LDA_U32];
  __shared__ unsigned sBzp[2][BLK_N][LDB_U32];
  __shared__ unsigned sBhp[2][BLK_N][LDB_U32];

  const int tid  = threadIdx.x;
  const int lane = tid & 31;
  const int wave = tid >> 5;
  const int wm   = wave >> 1;   // 0..3
  const int wn   = wave & 1;    // 0..1
  const int r    = lane & 15;
  const int hi   = lane >> 4;

  const int n0 = blockIdx.x * BLK_N;
  const int m0 = blockIdx.y * BLK_M;

  v8f accZ[2][2], accH[2][2];
  #pragma unroll
  for (int mt = 0; mt < 2; ++mt)
    #pragma unroll
    for (int nt = 0; nt < 2; ++nt) { accZ[mt][nt] = (v8f){}; accH[mt][nt] = (v8f){}; }

  // staging maps
  const int arow = tid >> 1;            // 0..127
  const int acol = (tid & 1) * 16;      // 0 or 16 (f32 col base)
  const int bkp  = tid >> 4;            // 0..15 : K pair index
  const int bnc  = (tid & 15) * 4;      // 0..60 : N col base

  float4 ax[4], vz0, vz1, vh0, vh1;

  auto glb_load = [&](int k0) {
    #pragma unroll
    for (int j = 0; j < 4; ++j)
      ax[j] = *(const float4*)&x[(size_t)(m0 + arow) * D_DIM + (k0 + acol + 4 * j)];
    vz0 = *(const float4*)&Wz[(size_t)(k0 + 2 * bkp)     * H_DIM + n0 + bnc];
    vz1 = *(const float4*)&Wz[(size_t)(k0 + 2 * bkp + 1) * H_DIM + n0 + bnc];
    vh0 = *(const float4*)&Wh[(size_t)(k0 + 2 * bkp)     * H_DIM + n0 + bnc];
    vh1 = *(const float4*)&Wh[(size_t)(k0 + 2 * bkp + 1) * H_DIM + n0 + bnc];
  };
  auto lds_store = [&](int buf) {
    #pragma unroll
    for (int j = 0; j < 4; ++j) {
      uint2 p;
      p.x = pack_bf16(ax[j].x, ax[j].y);
      p.y = pack_bf16(ax[j].z, ax[j].w);
      *(uint2*)&sAp[buf][arow][acol / 2 + 2 * j] = p;
    }
    sBzp[buf][bnc + 0][bkp] = pack_bf16(vz0.x, vz1.x);
    sBzp[buf][bnc + 1][bkp] = pack_bf16(vz0.y, vz1.y);
    sBzp[buf][bnc + 2][bkp] = pack_bf16(vz0.z, vz1.z);
    sBzp[buf][bnc + 3][bkp] = pack_bf16(vz0.w, vz1.w);
    sBhp[buf][bnc + 0][bkp] = pack_bf16(vh0.x, vh1.x);
    sBhp[buf][bnc + 1][bkp] = pack_bf16(vh0.y, vh1.y);
    sBhp[buf][bnc + 2][bkp] = pack_bf16(vh0.z, vh1.z);
    sBhp[buf][bnc + 3][bkp] = pack_bf16(vh0.w, vh1.w);
  };

  glb_load(0);
  lds_store(0);
  int bufi = 0;

  #pragma unroll 2
  for (int kt = 0; kt < NK; ++kt) {
    if (kt + 1 < NK) glb_load((kt + 1) * BLK_K);
    __syncthreads();   // current buffer's stores visible

    v16bf af[2];
    #pragma unroll
    for (int mt = 0; mt < 2; ++mt)
      af[mt] = load_frag_a(&sAp[bufi][wm * 32 + mt * 16 + r][hi * 8]);

    #pragma unroll
    for (int nt = 0; nt < 2; ++nt) {
      const int col = wn * 32 + nt * 16 + r;
      const v16bf bzf = load_frag_b(&sBzp[bufi][col][hi * 8]);
      const v16bf bhf = load_frag_b(&sBhp[bufi][col][hi * 8]);
      #pragma unroll
      for (int mt = 0; mt < 2; ++mt) {
        accZ[mt][nt] = __builtin_amdgcn_wmma_f32_16x16x32_bf16(
            false, af[mt], false, bzf, (short)0, accZ[mt][nt], false, false);
        accH[mt][nt] = __builtin_amdgcn_wmma_f32_16x16x32_bf16(
            false, af[mt], false, bhf, (short)0, accH[mt][nt], false, false);
      }
    }
    __syncthreads();   // all reads of this buffer complete
    if (kt + 1 < NK) lds_store(bufi ^ 1);
    bufi ^= 1;
  }

  // ---- epilogue: bias + log-space transforms ----
  #pragma unroll
  for (int nt = 0; nt < 2; ++nt) {
    const int n = n0 + wn * 32 + nt * 16 + r;
    const float bzv = bz[n];
    const float bhv = bh[n];
    #pragma unroll
    for (int mt = 0; mt < 2; ++mt) {
      #pragma unroll
      for (int vr = 0; vr < 8; ++vr) {
        const int m = m0 + wm * 32 + mt * 16 + hi * 8 + vr;
        const float kz = accZ[mt][nt][vr] + bzv;
        const float th = accH[mt][nt][vr] + bhv;
        const float log_z  = -softplusf(-kz);
        const float log_c  = -softplusf(kz);
        const float log_th = (th >= 0.0f) ? __logf(th + 0.5f) : -softplusf(-th);
        const size_t idx = (size_t)m * H_DIM + n;
        la[idx] = log_c;
        lb[idx] = log_z + log_th;
      }
    }
  }
}

// ---------------------------------------------------------------------------
// Kernel 2: per-channel log-space recurrence (== Heinsen scan), streamed via
// the Tensor Data Mover: wave 0 issues 2-D TDM tile loads (256 ch x SCAN_T
// steps) for la/lb into double-buffered LDS; all waves consume after
// s_wait_tensorcnt + barrier.
// ---------------------------------------------------------------------------
#define SCAN_T 8
__global__ __launch_bounds__(256)
void mingru_scan_kernel(const float* __restrict__ la,
                        float* __restrict__ out,
                        const float* __restrict__ h0) {
  __shared__ float sLa[2][SCAN_T][256];
  __shared__ float sLb[2][SCAN_T][256];
  const int tid  = threadIdx.x;
  const int t    = blockIdx.x * 256 + tid;
  const int hcol = (blockIdx.x & 3) * 256;     // H/256 = 4 blocks per batch
  const int bb   = blockIdx.x >> 2;

  float running = __logf(h0[t]);
  const float* laBase  = la  + (size_t)bb * S_DIM * H_DIM + hcol;
  float*       outBase = out + (size_t)bb * S_DIM * H_DIM + hcol;

  const bool issuer = (tid < 32);              // wave 0 only (uniform per wave)
  if (issuer) {
    tdm_load_2d(lds_offset(&sLa[0][0][0]), laBase,  256, SCAN_T, H_DIM);
    tdm_load_2d(lds_offset(&sLb[0][0][0]), outBase, 256, SCAN_T, H_DIM);
  }
  int bufi = 0;
  const int NC = S_DIM / SCAN_T;
  for (int c = 0; c < NC; ++c) {
    if (issuer) __builtin_amdgcn_s_wait_tensorcnt(0);
    __syncthreads();                           // chunk c resident in LDS
    if (issuer && (c + 1 < NC)) {
      const size_t off = (size_t)(c + 1) * SCAN_T * H_DIM;
      tdm_load_2d(lds_offset(&sLa[bufi ^ 1][0][0]), laBase + off,  256, SCAN_T, H_DIM);
      tdm_load_2d(lds_offset(&sLb[bufi ^ 1][0][0]), outBase + off, 256, SCAN_T, H_DIM);
    }
    size_t gidx = (size_t)c * SCAN_T * H_DIM + tid;
    #pragma unroll
    for (int j = 0; j < SCAN_T; ++j) {
      const float a  = sLa[bufi][j][tid];
      const float bv = sLb[bufi][j][tid];
      const float u  = a + running;
      const float mx = fmaxf(u, bv);
      const float mn = fminf(u, bv);
      running = mx + log1pf(__expf(mn - mx));
      outBase[gidx] = __expf(running);
      gidx += H_DIM;
    }
    __syncthreads();                           // buffer free for chunk c+2
    bufi ^= 1;
  }
}

// ---------------------------------------------------------------------------
extern "C" void kernel_launch(void* const* d_in, const int* in_sizes, int n_in,
                              void* d_out, int out_size, void* d_ws, size_t ws_size,
                              hipStream_t stream) {
  const float* x  = (const float*)d_in[0];
  const float* h0 = (const float*)d_in[1];
  const float* Wz = (const float*)d_in[2];
  const float* bz = (const float*)d_in[3];
  const float* Wh = (const float*)d_in[4];
  const float* bh = (const float*)d_in[5];
  float* out = (float*)d_out;
  float* la  = (float*)d_ws;   // B*S*H f32 = 128 MiB of scratch

  dim3 grid1(H_DIM / BLK_N, M_TOTAL / BLK_M);  // (16, 256)
  mingru_gemm_kernel<<<grid1, 256, 0, stream>>>(x, Wz, Wh, bz, bh, la, out);

  const int nchan = B_DIM * H_DIM;             // 8192
  mingru_scan_kernel<<<nchan / 256, 256, 0, stream>>>(la, out, h0);
}
